// Multi_head_Attention_67439576481949
// MI455X (gfx1250) — matrix operand order repaired and checked by reading, not compiled, not run on hardware
//
#include <hip/hip_runtime.h>
#include <hip/hip_bf16.h>
#include <stdint.h>
#include <stddef.h>

// ---------------------------------------------------------------------------
// Multi-head attention for MI455X (gfx1250, wave32, WMMA).
// Pipeline: convert->bf16, QKV GEMM (WMMA), RoPE+split, flash-attention
// (WMMA scores + WMMA attn*V with online softmax), output GEMM (WMMA).
// ---------------------------------------------------------------------------

typedef __attribute__((ext_vector_type(16))) __bf16 v16bf;
typedef __attribute__((ext_vector_type(8)))  float  v8f;

#define B_   8
#define L_   1024
#define DM_  768
#define NH_  12
#define HD_  64
#define ML_  (B_ * L_)          // 8192 rows
#define NEG_BIG (-1.0e30f)

// ---- helpers --------------------------------------------------------------

static __device__ __forceinline__ __bf16 f2bf(float f) {
  unsigned int u = __builtin_bit_cast(unsigned int, f);
  u += 0x7FFFu + ((u >> 16) & 1u);            // round-to-nearest-even
  unsigned short h = (unsigned short)(u >> 16);
  return __builtin_bit_cast(__bf16, h);
}

// place a packed pair of bf16 (from one dword load) into frag[2v], frag[2v+1]
static __device__ __forceinline__ void put2(v16bf& f, int v, unsigned int u) {
  f[2 * v]     = __builtin_bit_cast(__bf16, (unsigned short)(u & 0xFFFFu));
  f[2 * v + 1] = __builtin_bit_cast(__bf16, (unsigned short)(u >> 16));
}

static __device__ __forceinline__ v8f wmma_bf16(v16bf a, v16bf b, v8f c) {
  // D(16x16,f32) = A(16x32,bf16) * B(32x16,bf16) + C
  return __builtin_amdgcn_wmma_f32_16x16x32_bf16(
      false, a, false, b, (short)0, c, false, false);
}

static __device__ __forceinline__ float redmax16(float x) {
  x = fmaxf(x, __shfl_xor(x, 1));
  x = fmaxf(x, __shfl_xor(x, 2));
  x = fmaxf(x, __shfl_xor(x, 4));
  x = fmaxf(x, __shfl_xor(x, 8));
  return x;   // reduced over 16-lane half-wave (rows stay separate)
}
static __device__ __forceinline__ float redsum16(float x) {
  x += __shfl_xor(x, 1);
  x += __shfl_xor(x, 2);
  x += __shfl_xor(x, 4);
  x += __shfl_xor(x, 8);
  return x;
}

// ---- 1) fp32 -> bf16 conversion ------------------------------------------

__global__ void convert_bf16_kernel(const float* __restrict__ src,
                                    __bf16* __restrict__ dst, int n) {
  int i = blockIdx.x * blockDim.x + threadIdx.x;
  int stride = gridDim.x * blockDim.x;
  for (; i < n; i += stride) dst[i] = f2bf(src[i]);
}

// ---- 2) generic bf16 WMMA GEMM: C[M,N] = A[M,K] * Bt[N,K]^T + bias[N] ----
// Block = 128 threads (4 waves), each block computes a 64x64 tile.
// Wave w owns rows [m0, m0+16); 4 column tiles of 16.

__global__ void __launch_bounds__(128) gemm_bf16_kernel(
    const __bf16* __restrict__ A, const __bf16* __restrict__ Bt,
    const float* __restrict__ bias, float* __restrict__ C,
    int M, int N, int K) {
  const int wave = threadIdx.x >> 5;
  const int lane = threadIdx.x & 31;
  const int ln   = lane & 15;
  const int hi   = lane >> 4;
  const int m0   = blockIdx.x * 64 + wave * 16;
  const int n0   = blockIdx.y * 64;

  v8f zf = {0.f, 0.f, 0.f, 0.f, 0.f, 0.f, 0.f, 0.f};
  v8f acc[4];
#pragma unroll
  for (int t = 0; t < 4; ++t) acc[t] = zf;

  for (int k0 = 0; k0 < K; k0 += 32) {
    // A fragment: row m = ln, K index = hi*16 + 2v + h
    v16bf Af;
    const unsigned int* arow =
        (const unsigned int*)(A + (size_t)(m0 + ln) * K + k0);
#pragma unroll
    for (int v = 0; v < 8; ++v) {
      int kk = hi * 16 + 2 * v;
      put2(Af, v, arow[kk >> 1]);
    }
#pragma unroll
    for (int t = 0; t < 4; ++t) {
      // B fragment: col n = ln, K index = hi*16 + 2v + h
      v16bf Bf;
      const unsigned int* brow =
          (const unsigned int*)(Bt + (size_t)(n0 + t * 16 + ln) * K + k0);
#pragma unroll
      for (int v = 0; v < 8; ++v) put2(Bf, v, brow[hi * 8 + v]);
      acc[t] = wmma_bf16(Af, Bf, acc[t]);
    }
  }

  // D layout: element (m = v + 8*hi, n = ln)
#pragma unroll
  for (int t = 0; t < 4; ++t) {
#pragma unroll
    for (int v = 0; v < 8; ++v) {
      int row = m0 + v + 8 * hi;
      int col = n0 + t * 16 + ln;
      C[(size_t)row * N + col] = acc[t][v] + bias[col];
    }
  }
}

// ---- 3) RoPE + split into head-major bf16 q/k/v [B, NH, L, HD] ------------

__global__ void rope_split_kernel(const float* __restrict__ qkv,
                                  __bf16* __restrict__ qb,
                                  __bf16* __restrict__ kb,
                                  __bf16* __restrict__ vb) {
  int i = blockIdx.x * blockDim.x + threadIdx.x;
  const int total = B_ * L_ * NH_ * (HD_ / 2);
  if (i >= total) return;
  int p  = i & 31;          // rotation pair index (0..31)
  int t  = i >> 5;
  int nh = t % NH_;
  int t2 = t / NH_;
  int l  = t2 & (L_ - 1);
  int b  = t2 >> 10;

  size_t base = (size_t)(b * L_ + l) * (3 * DM_);
  int off = nh * HD_ + 2 * p;
  float q1 = qkv[base + off],            q2 = qkv[base + off + 1];
  float k1 = qkv[base + DM_ + off],      k2 = qkv[base + DM_ + off + 1];
  float v1 = qkv[base + 2 * DM_ + off],  v2 = qkv[base + 2 * DM_ + off + 1];

  float fi  = (float)(2 * p) * (1.0f / (float)HD_);
  float inv = __expf(-9.2103403719761836f * fi);   // 10000^(-2p/D)
  float ang = (float)l * inv;
  float s, c;
  __sincosf(ang, &s, &c);

  size_t dst = ((size_t)(b * NH_ + nh) * L_ + l) * HD_ + 2 * p;
  qb[dst]     = f2bf(q1 * c - q2 * s);
  qb[dst + 1] = f2bf(q2 * c + q1 * s);
  kb[dst]     = f2bf(k1 * c - k2 * s);
  kb[dst + 1] = f2bf(k2 * c + k1 * s);
  vb[dst]     = f2bf(v1);
  vb[dst + 1] = f2bf(v2);
}

// ---- 4) flash attention: per wave 16 q-rows, stream keys 32 at a time -----

__global__ void __launch_bounds__(128) attention_kernel(
    const __bf16* __restrict__ qbuf, const __bf16* __restrict__ kbuf,
    const __bf16* __restrict__ vbuf, const float* __restrict__ pos_bias,
    const int* __restrict__ mask, __bf16* __restrict__ obuf) {
  __shared__ float plds[4][16 * 32];   // per-wave P transpose staging

  const int wave = threadIdx.x >> 5;
  const int lane = threadIdx.x & 31;
  const int ln   = lane & 15;
  const int hi   = lane >> 4;
  const int b    = blockIdx.z;
  const int h    = blockIdx.y;
  const int q0   = blockIdx.x * 64 + wave * 16;

  const __bf16* qp = qbuf + (size_t)(b * NH_ + h) * L_ * HD_;
  const __bf16* kp = kbuf + (size_t)(b * NH_ + h) * L_ * HD_;
  const __bf16* vp = vbuf + (size_t)(b * NH_ + h) * L_ * HD_;
  const float*  pb = pos_bias + (size_t)h * L_ * L_;
  const int*    mk = mask + b * L_;
  float* pl = &plds[wave][0];

  // Q A-fragments: Qfa covers d=0..31, Qfb covers d=32..63
  v16bf Qfa, Qfb;
  {
    const unsigned int* qrow =
        (const unsigned int*)(qp + (size_t)(q0 + ln) * HD_);
#pragma unroll
    for (int v = 0; v < 8; ++v) {
      int kk = hi * 16 + 2 * v;
      put2(Qfa, v, qrow[kk >> 1]);
      put2(Qfb, v, qrow[(kk + 32) >> 1]);
    }
  }

  v8f zf = {0.f, 0.f, 0.f, 0.f, 0.f, 0.f, 0.f, 0.f};
  v8f O[4];
#pragma unroll
  for (int t = 0; t < 4; ++t) O[t] = zf;
  float rmax[8], rsum[8];
#pragma unroll
  for (int v = 0; v < 8; ++v) { rmax[v] = NEG_BIG; rsum[v] = 0.0f; }

  for (int k0 = 0; k0 < L_; k0 += 32) {
    if (k0 + 32 < L_) {
      __builtin_prefetch(kp + (size_t)(k0 + 32 + ln) * HD_, 0, 0);
      __builtin_prefetch(vp + (size_t)(k0 + 32 + ln) * HD_, 0, 0);
    }
    float mb0 = (mk[k0 + ln] > 0)      ? 0.0f : NEG_BIG;
    float mb1 = (mk[k0 + 16 + ln] > 0) ? 0.0f : NEG_BIG;

    // ---- scores: two 16x16 tiles (keys k0..+15 and k0+16..+31) ----
    v8f S0 = zf, S1 = zf;
    {
      v16bf Blo, Bhi;
      const unsigned int* krow =
          (const unsigned int*)(kp + (size_t)(k0 + ln) * HD_);
#pragma unroll
      for (int v = 0; v < 8; ++v) {
        put2(Blo, v, krow[hi * 8 + v]);        // d = hi*16 + 2v (+h)
        put2(Bhi, v, krow[16 + hi * 8 + v]);   // d += 32
      }
      S0 = wmma_bf16(Qfa, Blo, S0);
      S0 = wmma_bf16(Qfb, Bhi, S0);
      krow = (const unsigned int*)(kp + (size_t)(k0 + 16 + ln) * HD_);
#pragma unroll
      for (int v = 0; v < 8; ++v) {
        put2(Blo, v, krow[hi * 8 + v]);
        put2(Bhi, v, krow[16 + hi * 8 + v]);
      }
      S1 = wmma_bf16(Qfa, Blo, S1);
      S1 = wmma_bf16(Qfb, Bhi, S1);
    }

    // ---- bias + online softmax (per row m = v + 8*hi) ----
    float p0[8], p1[8];
#pragma unroll
    for (int v = 0; v < 8; ++v) {
      int m = v + 8 * hi;
      const float* pr = pb + (size_t)(q0 + m) * L_ + k0;
      float s0 = S0[v] * 0.125f + pr[ln] + mb0;        // 1/sqrt(64)=0.125
      float s1 = S1[v] * 0.125f + pr[16 + ln] + mb1;
      float bm = redmax16(fmaxf(s0, s1));
      float nm = fmaxf(rmax[v], bm);
      float alpha = __expf(rmax[v] - nm);
      rmax[v] = nm;
      float e0 = __expf(s0 - nm);
      float e1 = __expf(s1 - nm);
      rsum[v] = rsum[v] * alpha + redsum16(e0 + e1);
      p0[v] = e0;
      p1[v] = e1;
#pragma unroll
      for (int t = 0; t < 4; ++t) O[t][v] *= alpha;
    }

    // ---- P: D-layout -> LDS -> A-layout (bf16) ----
#pragma unroll
    for (int v = 0; v < 8; ++v) {
      int m = v + 8 * hi;
      pl[m * 32 + ln]      = p0[v];
      pl[m * 32 + 16 + ln] = p1[v];
    }
    __builtin_amdgcn_wave_barrier();     // wave-private region; LDS is
    asm volatile("" ::: "memory");       // in-order per wave
    v16bf Pf;
#pragma unroll
    for (int v = 0; v < 8; ++v) {
      int kk = hi * 16 + 2 * v;
      Pf[2 * v]     = f2bf(pl[ln * 32 + kk]);
      Pf[2 * v + 1] = f2bf(pl[ln * 32 + kk + 1]);
    }
    __builtin_amdgcn_wave_barrier();
    asm volatile("" ::: "memory");

    // ---- O[16 x 64] += P[16 x 32] * V[32 x 64] (4 column tiles) ----
#pragma unroll
    for (int t = 0; t < 4; ++t) {
      v16bf Vf;
#pragma unroll
      for (int v = 0; v < 8; ++v) {
        int kk = hi * 16 + 2 * v;        // key index within block
        Vf[2 * v]     = vp[(size_t)(k0 + kk)     * HD_ + t * 16 + ln];
        Vf[2 * v + 1] = vp[(size_t)(k0 + kk + 1) * HD_ + t * 16 + ln];
      }
      O[t] = wmma_bf16(Pf, Vf, O[t]);
    }
  }

  // ---- normalize and store o[b, l, h*64 + d] as bf16 ----
#pragma unroll
  for (int v = 0; v < 8; ++v) {
    int m = v + 8 * hi;
    float inv = 1.0f / rsum[v];
#pragma unroll
    for (int t = 0; t < 4; ++t) {
      obuf[(size_t)(b * L_ + q0 + m) * DM_ + h * HD_ + t * 16 + ln] =
          f2bf(O[t][v] * inv);
    }
  }
}

// ---- host-side orchestration ---------------------------------------------
// Workspace layout (bytes):                          offset        size
//   x_bf16   [8192 x 768]  bf16                             0  12,582,912
//   wq_bf16  [2304 x 768]  bf16                    12,582,912   3,538,944
//   wo_bf16  [ 768 x 768]  bf16                    16,121,856   1,179,648
//   qkv_f32  [8192 x 2304] f32                     17,301,504  75,497,472
//   q_bf16   [8,12,1024,64] bf16                   92,798,976  12,582,912
//   k_bf16                                        105,381,888  12,582,912
//   v_bf16                                        117,964,800  12,582,912
//   o_bf16   [8192 x 768]  bf16                   130,547,712  12,582,912
//   total: 143,130,624 bytes

extern "C" void kernel_launch(void* const* d_in, const int* in_sizes, int n_in,
                              void* d_out, int out_size, void* d_ws,
                              size_t ws_size, hipStream_t stream) {
  (void)in_sizes; (void)n_in; (void)out_size; (void)ws_size;
  const float* x     = (const float*)d_in[0];
  const float* posb  = (const float*)d_in[1];
  const float* qkv_w = (const float*)d_in[2];
  const float* qkv_b = (const float*)d_in[3];
  const float* out_w = (const float*)d_in[4];
  const float* out_b = (const float*)d_in[5];
  const int*   mask  = (const int*)d_in[6];
  float* out = (float*)d_out;

  char* ws = (char*)d_ws;
  __bf16* x_bf  = (__bf16*)(ws);
  __bf16* wq_bf = (__bf16*)(ws + 12582912);
  __bf16* wo_bf = (__bf16*)(ws + 16121856);
  float*  qkv   = (float*)(ws + 17301504);
  __bf16* qbuf  = (__bf16*)(ws + 92798976);
  __bf16* kbuf  = (__bf16*)(ws + 105381888);
  __bf16* vbuf  = (__bf16*)(ws + 117964800);
  __bf16* obuf  = (__bf16*)(ws + 130547712);

  convert_bf16_kernel<<<2048, 256, 0, stream>>>(x, x_bf, ML_ * DM_);
  convert_bf16_kernel<<<2048, 256, 0, stream>>>(qkv_w, wq_bf, 3 * DM_ * DM_);
  convert_bf16_kernel<<<1024, 256, 0, stream>>>(out_w, wo_bf, DM_ * DM_);

  // qkv = x @ qkv_w^T + qkv_b : [8192 x 2304]
  gemm_bf16_kernel<<<dim3(ML_ / 64, (3 * DM_) / 64), 128, 0, stream>>>(
      x_bf, wq_bf, qkv_b, qkv, ML_, 3 * DM_, DM_);

  rope_split_kernel<<<(B_ * L_ * NH_ * (HD_ / 2) + 255) / 256, 256, 0,
                      stream>>>(qkv, qbuf, kbuf, vbuf);

  attention_kernel<<<dim3(L_ / 64, NH_, B_), 128, 0, stream>>>(
      qbuf, kbuf, vbuf, posb, mask, obuf);

  // out = o @ out_w^T + out_b : [8192 x 768]
  gemm_bf16_kernel<<<dim3(ML_ / 64, DM_ / 64), 128, 0, stream>>>(
      obuf, wo_bf, out_b, out, ML_, DM_, DM_);
}
